// HomogeneousGNN_19155554140462
// MI455X (gfx1250) — matrix operand order repaired and checked by reading, not compiled, not run on hardware
//
#include <hip/hip_runtime.h>

#define NN 50000
#define NE 800000
#define DIN 128
#define HID 256
#define DOUT 128

typedef __attribute__((ext_vector_type(16))) __bf16 v16bf;
typedef __attribute__((ext_vector_type(8)))  __bf16 v8bf;
typedef __attribute__((ext_vector_type(8)))  float  v8f;

// matches the builtin's parameter type: 'int __attribute__((vector_size(16)))'
typedef int v4i __attribute__((vector_size(16)));
typedef __attribute__((address_space(3))) v4i lds_v4i;

#if defined(__has_builtin)
#if __has_builtin(__builtin_amdgcn_global_load_async_to_lds_b128) && \
    __has_builtin(__builtin_amdgcn_s_wait_asynccnt)
#define HAS_ASYNC 1
#endif
#endif
#ifndef HAS_ASYNC
#define HAS_ASYNC 0
#endif

// ---------- utility ----------
__global__ void k_zero(float* __restrict__ p, long long n) {
  long long i = (long long)blockIdx.x * blockDim.x + threadIdx.x;
  if (i < n) p[i] = 0.0f;
}

__global__ void k_count(const long long* __restrict__ dst, float* __restrict__ cnt) {
  int e = blockIdx.x * blockDim.x + threadIdx.x;
  if (e < NE) atomicAdd(&cnt[dst[e]], 1.0f);
}

__global__ void k_inv(const float* __restrict__ cnt, float* __restrict__ inv) {
  int i = blockIdx.x * blockDim.x + threadIdx.x;
  if (i < NN) inv[i] = 1.0f / fmaxf(cnt[i], 1.0f);
}

// ---------- scatter-add: one wave per edge, coalesced f32 atomics ----------
__global__ void k_scatter(const float* __restrict__ x, const long long* __restrict__ src,
                          const long long* __restrict__ dst, float* __restrict__ agg, int D) {
  int e = blockIdx.x * 8 + (threadIdx.x >> 5);   // 8 waves / block
  if (e >= NE) return;
  int lane = threadIdx.x & 31;
  long long s = src[e] * (long long)D;
  long long d = dst[e] * (long long)D;
  for (int f = lane; f < D; f += 32)
    atomicAdd(&agg[d + f], x[s + f]);
}

// ---------- mean + pack A = [mean || x] in bf16, row-major K = 2D ----------
__global__ void k_finalize(const float* __restrict__ agg, const float* __restrict__ xin,
                           const float* __restrict__ inv, __bf16* __restrict__ A, int D) {
  long long idx = (long long)blockIdx.x * blockDim.x + threadIdx.x;
  long long total = (long long)NN * D;
  if (idx >= total) return;
  int i = (int)(idx / D);
  int f = (int)(idx % D);
  long long K2 = 2LL * D;
  A[(long long)i * K2 + f]     = (__bf16)(agg[idx] * inv[i]);
  A[(long long)i * K2 + D + f] = (__bf16)xin[idx];
}

// ---------- pack Wt[n][k] = concat(W_l, W_r)[k][n] in bf16 ----------
__global__ void k_prepw(const float* __restrict__ Wl, const float* __restrict__ Wr,
                        __bf16* __restrict__ Wt, int D, int N) {
  int K2 = 2 * D;
  long long idx = (long long)blockIdx.x * blockDim.x + threadIdx.x;
  long long total = (long long)K2 * N;
  if (idx >= total) return;
  int n = (int)(idx / K2);
  int k = (int)(idx % K2);
  float v = (k < D) ? Wl[(long long)k * N + n] : Wr[(long long)(k - D) * N + n];
  Wt[idx] = (__bf16)v;
}

// ---------- WMMA GEMM: C[M,N] = A[M,K] @ Wt^T + bias, optional ReLU ----------
// block = 256 threads (8 waves). Block tile 128x64. Wave w: rows [w*16, w*16+16) x 64 cols.
// A tile double-buffered in LDS; staged via GLOBAL_LOAD_ASYNC_TO_LDS_B128.
// All waves compute unconditionally (A rows clamped to M-1 during staging, so LDS is
// always valid); only the epilogue stores are guarded. Keeps EXEC all-ones through WMMA
// and avoids accumulator copies around an exec-masked region.
__global__ __launch_bounds__(256) void
k_gemm(const __bf16* __restrict__ A, const __bf16* __restrict__ Wt,
       const float* __restrict__ bias, float* __restrict__ C,
       int M, int N, int K, int relu) {
  __shared__ __align__(16) __bf16 As[2][128 * 40];  // pitch 40 halves -> conflict-free frags

  const int tid  = threadIdx.x;
  const int w    = tid >> 5;
  const int lane = tid & 31;
  const int r    = lane & 15;       // row within A tile / column within B tile
  const int kg   = lane >> 4;       // K-group select (0/1)
  const int m0   = blockIdx.x * 128;
  const int n0   = blockIdx.y * 64;

  // stage 128 rows x 32 bf16 of A (cols [kb_, kb_+32)) into LDS buffer buf_.
  // OOB rows clamp to M-1 (their results are discarded by the store guard).
  auto stage = [&](int kb_, int buf_) {
#pragma unroll
    for (int i = 0; i < 2; ++i) {
      int idx = tid * 2 + i;        // 0..511 chunks of 16B
      int row = idx >> 2;           // 0..127
      int c16 = idx & 3;
      int gr  = m0 + row;
      if (gr > M - 1) gr = M - 1;
#if HAS_ASYNC
      __builtin_amdgcn_global_load_async_to_lds_b128(
          (v4i*)(A + (long long)gr * K + kb_ + c16 * 8),
          (lds_v4i*)&As[buf_][row * 40 + c16 * 8], 0, 0);
#else
      v8bf v = *(const v8bf*)(A + (long long)gr * K + kb_ + c16 * 8);
      *(v8bf*)(&As[buf_][row * 40 + c16 * 8]) = v;
#endif
    }
  };
  auto wait_stage = [&]() {
#if HAS_ASYNC
    __builtin_amdgcn_s_wait_asynccnt(0);
#endif
    __syncthreads();
  };

  v8f acc[4] = {};

  stage(0, 0);
  wait_stage();

  int buf = 0;
  for (int kb = 0; kb < K; kb += 32) {
    if (kb + 32 < K) stage(kb + 32, buf ^ 1);   // async prefetch next k-block

    // A fragment: lanes 0-15 -> K {0..15}, lanes 16-31 -> K {16..31}
    int lr = w * 16 + r;
    v8bf lo = *(const v8bf*)(&As[buf][lr * 40 + kg * 16]);
    v8bf hi = *(const v8bf*)(&As[buf][lr * 40 + kg * 16 + 8]);
    v16bf a = __builtin_shufflevector(lo, hi, 0,1,2,3,4,5,6,7,8,9,10,11,12,13,14,15);
#pragma unroll
    for (int nt = 0; nt < 4; ++nt) {
      // B fragment: column n = n0+nt*16+r, 16 consecutive K at kb + kg*16 (32B contiguous)
      const __bf16* bp = Wt + (long long)(n0 + nt * 16 + r) * K + kb + kg * 16;
      v16bf b = *(const v16bf*)bp;
      acc[nt] = __builtin_amdgcn_wmma_f32_16x16x32_bf16(
          false, a, false, b, (short)0, acc[nt], false, false);
    }

    wait_stage();
    buf ^= 1;
  }

  if ((m0 + w * 16) < M) {
    // C layout: VGPR j -> M = j + kg*8; N = lane&15
    int mbase = m0 + w * 16 + kg * 8;
#pragma unroll
    for (int nt = 0; nt < 4; ++nt) {
      int col = n0 + nt * 16 + r;
      float bv = bias[col];
#pragma unroll
      for (int j = 0; j < 8; ++j) {
        float val = acc[nt][j] + bv;
        if (relu) val = fmaxf(val, 0.0f);
        C[(long long)(mbase + j) * N + col] = val;
      }
    }
  }
}

// ---------- host orchestration ----------
extern "C" void kernel_launch(void* const* d_in, const int* in_sizes, int n_in,
                              void* d_out, int out_size, void* d_ws, size_t ws_size,
                              hipStream_t stream) {
  const float*     x   = (const float*)d_in[0];
  const long long* ei  = (const long long*)d_in[1];
  const long long* src = ei;
  const long long* dst = ei + NE;
  const float* Wl1 = (const float*)d_in[2];
  const float* bl1 = (const float*)d_in[3];
  const float* Wr1 = (const float*)d_in[4];
  const float* Wl2 = (const float*)d_in[5];
  const float* bl2 = (const float*)d_in[6];
  const float* Wr2 = (const float*)d_in[7];
  float* out = (float*)d_out;

  char* ws = (char*)d_ws;
  size_t off = 0;
  auto alloc = [&](size_t bytes) -> void* {
    void* p = ws + off;
    off = (off + bytes + 255) & ~(size_t)255;
    return p;
  };
  float*  cnt  = (float*)alloc((size_t)NN * 4);
  float*  inv  = (float*)alloc((size_t)NN * 4);
  float*  agg  = (float*)alloc((size_t)NN * HID * 4);      // reused both layers
  float*  h    = (float*)alloc((size_t)NN * HID * 4);      // layer-1 activations (f32)
  __bf16* Abuf = (__bf16*)alloc((size_t)NN * 512 * 2);     // packed [mean||x] bf16
  __bf16* Wt   = (__bf16*)alloc((size_t)512 * 256 * 2);    // packed transposed weights

  auto cdiv = [](long long a, long long b) { return (int)((a + b - 1) / b); };
  const int T = 256;

  // degree counts (shared by both layers)
  k_zero<<<cdiv(NN, T), T, 0, stream>>>(cnt, NN);
  k_count<<<cdiv(NE, T), T, 0, stream>>>(dst, cnt);
  k_inv<<<cdiv(NN, T), T, 0, stream>>>(cnt, inv);

  // ---- layer 1: D=128 -> HID=256, ReLU ----
  k_zero<<<cdiv((long long)NN * DIN, T), T, 0, stream>>>(agg, (long long)NN * DIN);
  k_scatter<<<cdiv(NE, 8), T, 0, stream>>>(x, src, dst, agg, DIN);
  k_finalize<<<cdiv((long long)NN * DIN, T), T, 0, stream>>>(agg, x, inv, Abuf, DIN);
  k_prepw<<<cdiv(2LL * DIN * HID, T), T, 0, stream>>>(Wl1, Wr1, Wt, DIN, HID);
  {
    dim3 g(cdiv(NN, 128), HID / 64);
    k_gemm<<<g, T, 0, stream>>>(Abuf, Wt, bl1, h, NN, HID, 2 * DIN, 1);
  }

  // ---- layer 2: HID=256 -> OUT=128, no ReLU ----
  k_zero<<<cdiv((long long)NN * HID, T), T, 0, stream>>>(agg, (long long)NN * HID);
  k_scatter<<<cdiv(NE, 8), T, 0, stream>>>(h, src, dst, agg, HID);
  k_finalize<<<cdiv((long long)NN * HID, T), T, 0, stream>>>(agg, h, inv, Abuf, HID);
  k_prepw<<<cdiv(2LL * HID * DOUT, T), T, 0, stream>>>(Wl2, Wr2, Wt, HID, DOUT);
  {
    dim3 g(cdiv(NN, 128), DOUT / 64);
    k_gemm<<<g, T, 0, stream>>>(Abuf, Wt, bl2, out, NN, DOUT, 2 * HID, 0);
  }
}
